// TrajectoryPredictor_35854386987560
// MI455X (gfx1250) — matrix operand order repaired and checked by reading, not compiled, not run on hardware
//
#include <hip/hip_runtime.h>
#include <cstddef>
#include <cstdint>

// ---------------------------------------------------------------------------
// Seq2seq GRU (enc 2x50, dec 2x30) for MI455X / gfx1250.
// v2: LDS-staged activation tiles (shared across the 4 waves of a block) +
// 2 batch-row tiles per wave so every weight fragment feeds 2 WMMAs.
// Core op: v_wmma_f32_16x16x32_bf16, fused with the GRU gate nonlinearity.
// ---------------------------------------------------------------------------

#define BATCH   32768
#define HDIM    512
#define DDIM    3
#define THIST   50
#define FUT     30

typedef __attribute__((ext_vector_type(16))) __bf16 v16bf;
typedef __attribute__((ext_vector_type(8)))  __bf16 v8bf;
typedef __attribute__((ext_vector_type(8)))  float  v8f;

// ---- fragment loaders (contiguous k order, same in A and B, wave32) ---------------
// A (16x32 bf16, row-major src, ld = HDIM):
//   lane L: m = m0 + (L&15), kb = k0 + 16*(L>=16)
//   halves[0..15] = A[m][kb..kb+15]
__device__ __forceinline__ v16bf a_frag_from(const __bf16* A, int m0, int k0, int lane) {
    int m  = m0 + (lane & 15);
    int kb = k0 + ((lane >> 4) << 4);
    const __bf16* p = A + (size_t)m * HDIM + kb;
    v8bf lo = *(const v8bf*)p;          // 16B chunks (kb multiple of 8)
    v8bf hi = *(const v8bf*)(p + 8);
    return __builtin_shufflevector(lo, hi, 0,1,2,3,4,5,6,7,8,9,10,11,12,13,14,15);
}

// B (32x16 bf16): lane L holds column n = n0 + (L&15),
//   halves[i] = B[k0 + 16*(L>=16) + i][n].  With B[k][n] = W[n0+n][k]
//   (row-major W, ld = HDIM) -> 16 contiguous halves, one 32B load.
__device__ __forceinline__ v16bf load_b_frag(const __bf16* __restrict__ W,
                                             int n0, int k0, int lane) {
    int n  = n0 + (lane & 15);
    int kb = k0 + ((lane >> 4) << 4);
    return *(const v16bf*)(W + (size_t)n * HDIM + kb);  // 32B aligned
}

__device__ __forceinline__ float sigmoidf_(float x) {
    return 1.0f / (1.0f + __expf(-x));
}
__device__ __forceinline__ float tanhf_(float x) {
    float e = __expf(-2.0f * fabsf(x));
    float t = (1.0f - e) / (1.0f + e);
    return copysignf(t, x);
}

#define WMMA_BF16(A_, B_, C_) \
    __builtin_amdgcn_wmma_f32_16x16x32_bf16(false, (A_), false, (B_), (short)0, (C_), false, false)

// ---------------------------------------------------------------------------
// GRU cell, input dim = 3 (layer 0). h-path via WMMA, x-path scalar.
// Block = 128 thr (4 waves) covers 32 batch rows x 64 hidden cols.
// h tile staged in LDS; each wave: 2 m-tiles x 1 n-tile, 6 wmma / k-step.
// grid = (HDIM/64, BATCH/32)
// ---------------------------------------------------------------------------
__global__ void __launch_bounds__(128)
gru_cell_small(const float* __restrict__ xbase, int xstride,
               const float* __restrict__ Wih,           // (3H x 3) f32
               const float* __restrict__ b_ih,
               const float* __restrict__ b_hh,
               const __bf16* __restrict__ Whh,          // (3H x H) bf16
               const __bf16* __restrict__ h_in,         // (B x H) bf16
               float*  __restrict__ h_f32,              // (B x H) in/out
               __bf16* __restrict__ h_out)              // (B x H) bf16 out
{
    __shared__ __bf16 Hs[32 * HDIM];                    // 32 KB
    const int tid  = threadIdx.x;
    const int lane = tid & 31;
    const int wv   = tid >> 5;
    const int n0   = (blockIdx.x * 4 + wv) * 16;        // hidden-col tile
    const int m0   = blockIdx.y * 32;                   // batch-row tile (32 rows)

    {   // cooperative stage: 32 rows x 512 cols bf16 = 32 KB = 2048 x 16B
        const uint4* g = (const uint4*)(h_in + (size_t)m0 * HDIM);
        uint4* s = (uint4*)Hs;
        #pragma unroll
        for (int i = 0; i < 16; ++i) s[tid + i * 128] = g[tid + i * 128];
    }
    __syncthreads();

    v8f accH[3][2];
    #pragma unroll
    for (int g = 0; g < 3; ++g) { accH[g][0] = (v8f){}; accH[g][1] = (v8f){}; }

    #pragma unroll 2
    for (int k0 = 0; k0 < HDIM; k0 += 32) {
        v16bf ah0 = a_frag_from(Hs,  0, k0, lane);
        v16bf ah1 = a_frag_from(Hs, 16, k0, lane);
        #pragma unroll
        for (int g = 0; g < 3; ++g) {
            v16bf bh = load_b_frag(Whh, g * HDIM + n0, k0, lane);
            accH[g][0] = WMMA_BF16(ah0, bh, accH[g][0]);
            accH[g][1] = WMMA_BF16(ah1, bh, accH[g][1]);
        }
    }

    const int j = n0 + (lane & 15);
    const float bi0 = b_ih[j], bi1 = b_ih[HDIM + j], bi2 = b_ih[2 * HDIM + j];
    const float bh0 = b_hh[j], bh1 = b_hh[HDIM + j], bh2 = b_hh[2 * HDIM + j];
    const float wr0 = Wih[(size_t)j * 3 + 0], wr1 = Wih[(size_t)j * 3 + 1], wr2 = Wih[(size_t)j * 3 + 2];
    const float wz0 = Wih[(size_t)(HDIM + j) * 3 + 0], wz1 = Wih[(size_t)(HDIM + j) * 3 + 1], wz2 = Wih[(size_t)(HDIM + j) * 3 + 2];
    const float wn0 = Wih[(size_t)(2 * HDIM + j) * 3 + 0], wn1 = Wih[(size_t)(2 * HDIM + j) * 3 + 1], wn2 = Wih[(size_t)(2 * HDIM + j) * 3 + 2];

    #pragma unroll
    for (int mt = 0; mt < 2; ++mt) {
        const int mb = m0 + mt * 16 + ((lane >> 4) << 3);  // lanes 16-31: rows +8
        #pragma unroll
        for (int v = 0; v < 8; ++v) {
            const int m = mb + v;
            const float* xr = xbase + (size_t)m * xstride;
            const float x0 = xr[0], x1 = xr[1], x2 = xr[2];
            const float gir = x0 * wr0 + x1 * wr1 + x2 * wr2 + bi0;
            const float giz = x0 * wz0 + x1 * wz1 + x2 * wz2 + bi1;
            const float gin = x0 * wn0 + x1 * wn1 + x2 * wn2 + bi2;
            const float r = sigmoidf_(gir + accH[0][mt][v] + bh0);
            const float z = sigmoidf_(giz + accH[1][mt][v] + bh1);
            const float n = tanhf_(gin + r * (accH[2][mt][v] + bh2));
            const size_t idx = (size_t)m * HDIM + j;
            const float hprev = h_f32[idx];
            const float hn = (1.0f - z) * n + z * hprev;
            h_f32[idx] = hn;
            h_out[idx] = (__bf16)hn;
        }
    }
}

// ---------------------------------------------------------------------------
// GRU cell, input dim = H (layer 1). Both paths via WMMA.
// Block covers 32 rows x 64 cols; X and H tiles staged in LDS (64 KB).
// Each wave: 12 wmma / k-step on 4 LDS A-frags + 6 global B-frags.
// ---------------------------------------------------------------------------
__global__ void __launch_bounds__(128)
gru_cell_big(const __bf16* __restrict__ Xbf,            // (B x H) activations
             const __bf16* __restrict__ Wih,            // (3H x H) bf16
             const __bf16* __restrict__ Whh,            // (3H x H) bf16
             const float* __restrict__ b_ih,
             const float* __restrict__ b_hh,
             const __bf16* __restrict__ h_in,           // (B x H) bf16
             float*  __restrict__ h_f32,
             __bf16* __restrict__ h_out)
{
    __shared__ __bf16 Xs[32 * HDIM];                    // 32 KB
    __shared__ __bf16 Hs[32 * HDIM];                    // 32 KB
    const int tid  = threadIdx.x;
    const int lane = tid & 31;
    const int wv   = tid >> 5;
    const int n0   = (blockIdx.x * 4 + wv) * 16;
    const int m0   = blockIdx.y * 32;

    {
        const uint4* gx = (const uint4*)(Xbf  + (size_t)m0 * HDIM);
        const uint4* gh = (const uint4*)(h_in + (size_t)m0 * HDIM);
        uint4* sx = (uint4*)Xs;
        uint4* sh = (uint4*)Hs;
        #pragma unroll
        for (int i = 0; i < 16; ++i) sx[tid + i * 128] = gx[tid + i * 128];
        #pragma unroll
        for (int i = 0; i < 16; ++i) sh[tid + i * 128] = gh[tid + i * 128];
    }
    __syncthreads();

    v8f accI[3][2], accH[3][2];
    #pragma unroll
    for (int g = 0; g < 3; ++g) {
        accI[g][0] = (v8f){}; accI[g][1] = (v8f){};
        accH[g][0] = (v8f){}; accH[g][1] = (v8f){};
    }

    #pragma unroll 2
    for (int k0 = 0; k0 < HDIM; k0 += 32) {
        v16bf ax0 = a_frag_from(Xs,  0, k0, lane);
        v16bf ax1 = a_frag_from(Xs, 16, k0, lane);
        v16bf ah0 = a_frag_from(Hs,  0, k0, lane);
        v16bf ah1 = a_frag_from(Hs, 16, k0, lane);
        #pragma unroll
        for (int g = 0; g < 3; ++g) {
            v16bf bi = load_b_frag(Wih, g * HDIM + n0, k0, lane);
            v16bf bh = load_b_frag(Whh, g * HDIM + n0, k0, lane);
            accI[g][0] = WMMA_BF16(ax0, bi, accI[g][0]);
            accI[g][1] = WMMA_BF16(ax1, bi, accI[g][1]);
            accH[g][0] = WMMA_BF16(ah0, bh, accH[g][0]);
            accH[g][1] = WMMA_BF16(ah1, bh, accH[g][1]);
        }
    }

    const int j = n0 + (lane & 15);
    const float bi0 = b_ih[j], bi1 = b_ih[HDIM + j], bi2 = b_ih[2 * HDIM + j];
    const float bh0 = b_hh[j], bh1 = b_hh[HDIM + j], bh2 = b_hh[2 * HDIM + j];

    #pragma unroll
    for (int mt = 0; mt < 2; ++mt) {
        const int mb = m0 + mt * 16 + ((lane >> 4) << 3);
        #pragma unroll
        for (int v = 0; v < 8; ++v) {
            const int m = mb + v;
            const float r = sigmoidf_(accI[0][mt][v] + bi0 + accH[0][mt][v] + bh0);
            const float z = sigmoidf_(accI[1][mt][v] + bi1 + accH[1][mt][v] + bh1);
            const float n = tanhf_(accI[2][mt][v] + bi2 + r * (accH[2][mt][v] + bh2));
            const size_t idx = (size_t)m * HDIM + j;
            const float hprev = h_f32[idx];
            const float hnew = (1.0f - z) * n + z * hprev;
            h_f32[idx] = hnew;
            h_out[idx] = (__bf16)hnew;
        }
    }
}

// ---------------------------------------------------------------------------
// pred = h @ fc_W.T + fc_b : [B,512]x[512,3]. One wave per batch row,
// lane-parallel over K with xor-shuffle reduction. Writes into d_out[b,t,:].
// ---------------------------------------------------------------------------
__global__ void __launch_bounds__(256)
fc_head(const float* __restrict__ h, const float* __restrict__ fcW,
        const float* __restrict__ fcb, float* __restrict__ out, int t)
{
    const int lane = threadIdx.x & 31;
    const int wv   = threadIdx.x >> 5;
    const int b    = blockIdx.x * 8 + wv;
    const float* hr = h + (size_t)b * HDIM;
    float s0 = 0.f, s1 = 0.f, s2 = 0.f;
    for (int k = lane; k < HDIM; k += 32) {
        const float v = hr[k];
        s0 += v * fcW[k];
        s1 += v * fcW[HDIM + k];
        s2 += v * fcW[2 * HDIM + k];
    }
    #pragma unroll
    for (int msk = 16; msk >= 1; msk >>= 1) {
        s0 += __shfl_xor(s0, msk, 32);
        s1 += __shfl_xor(s1, msk, 32);
        s2 += __shfl_xor(s2, msk, 32);
    }
    if (lane == 0) {
        float* o = out + (size_t)b * (FUT * DDIM) + (size_t)t * DDIM;
        o[0] = s0 + fcb[0];
        o[1] = s1 + fcb[1];
        o[2] = s2 + fcb[2];
    }
}

__global__ void __launch_bounds__(256)
cast_f32_to_bf16(const float* __restrict__ src, __bf16* __restrict__ dst, int n)
{
    int i = blockIdx.x * blockDim.x + threadIdx.x;
    if (i < n) dst[i] = (__bf16)src[i];
}

// ---------------------------------------------------------------------------
extern "C" void kernel_launch(void* const* d_in, const int* in_sizes, int n_in,
                              void* d_out, int out_size, void* d_ws, size_t ws_size,
                              hipStream_t stream)
{
    (void)in_sizes; (void)n_in; (void)out_size; (void)ws_size;

    const float* x_input   = (const float*)d_in[0];
    // d_in[1] = future_len (compile-time FUT)
    const float* enc_Wih0  = (const float*)d_in[2];
    const float* enc_Whh0  = (const float*)d_in[3];
    const float* enc_bih0  = (const float*)d_in[4];
    const float* enc_bhh0  = (const float*)d_in[5];
    const float* enc_Wih1  = (const float*)d_in[6];
    const float* enc_Whh1  = (const float*)d_in[7];
    const float* enc_bih1  = (const float*)d_in[8];
    const float* enc_bhh1  = (const float*)d_in[9];
    const float* dec_Wih0  = (const float*)d_in[10];
    const float* dec_Whh0  = (const float*)d_in[11];
    const float* dec_bih0  = (const float*)d_in[12];
    const float* dec_bhh0  = (const float*)d_in[13];
    const float* dec_Wih1  = (const float*)d_in[14];
    const float* dec_Whh1  = (const float*)d_in[15];
    const float* dec_bih1  = (const float*)d_in[16];
    const float* dec_bhh1  = (const float*)d_in[17];
    const float* fcW       = (const float*)d_in[18];
    const float* fcb       = (const float*)d_in[19];
    float* out = (float*)d_out;

    // ---- workspace carve-up ----
    char* ws = (char*)d_ws;
    const size_t BH    = (size_t)BATCH * HDIM;       // 16.78M elems
    const size_t F32S  = BH * sizeof(float);         // 64 MB
    const size_t BF16S = BH * sizeof(__bf16);        // 32 MB
    float*  h0f  = (float*)(ws);
    float*  h1f  = (float*)(ws + F32S);
    __bf16* h0bf[2] = { (__bf16*)(ws + 2 * F32S),
                        (__bf16*)(ws + 2 * F32S + BF16S) };
    __bf16* h1bf[2] = { (__bf16*)(ws + 2 * F32S + 2 * BF16S),
                        (__bf16*)(ws + 2 * F32S + 3 * BF16S) };
    char* wbase = ws + 2 * F32S + 4 * BF16S;
    const int    WN = 3 * HDIM * HDIM;               // 786432 elems per weight
    const size_t WB = (size_t)WN * sizeof(__bf16);   // 1.5 MB
    __bf16* encWhh0b = (__bf16*)(wbase + 0 * WB);
    __bf16* encWih1b = (__bf16*)(wbase + 1 * WB);
    __bf16* encWhh1b = (__bf16*)(wbase + 2 * WB);
    __bf16* decWhh0b = (__bf16*)(wbase + 3 * WB);
    __bf16* decWih1b = (__bf16*)(wbase + 4 * WB);
    __bf16* decWhh1b = (__bf16*)(wbase + 5 * WB);

    // ---- weight casts (L2-resident bf16 copies) ----
    const int cgrid = (WN + 255) / 256;
    cast_f32_to_bf16<<<cgrid, 256, 0, stream>>>(enc_Whh0, encWhh0b, WN);
    cast_f32_to_bf16<<<cgrid, 256, 0, stream>>>(enc_Wih1, encWih1b, WN);
    cast_f32_to_bf16<<<cgrid, 256, 0, stream>>>(enc_Whh1, encWhh1b, WN);
    cast_f32_to_bf16<<<cgrid, 256, 0, stream>>>(dec_Whh0, decWhh0b, WN);
    cast_f32_to_bf16<<<cgrid, 256, 0, stream>>>(dec_Wih1, decWih1b, WN);
    cast_f32_to_bf16<<<cgrid, 256, 0, stream>>>(dec_Whh1, decWhh1b, WN);

    // ---- h init = 0 (f32 and bf16 parity-0 buffers) ----
    hipMemsetAsync(h0f, 0, F32S, stream);
    hipMemsetAsync(h1f, 0, F32S, stream);
    hipMemsetAsync(h0bf[0], 0, BF16S, stream);
    hipMemsetAsync(h1bf[0], 0, BF16S, stream);

    const dim3 gblk(128);
    const dim3 ggrid(HDIM / 64, BATCH / 32);   // (8, 1024)
    int p = 0;

    // ---- encoder: 50 steps ----
    for (int t = 0; t < THIST; ++t) {
        gru_cell_small<<<ggrid, gblk, 0, stream>>>(
            x_input + (size_t)t * DDIM, THIST * DDIM,
            enc_Wih0, enc_bih0, enc_bhh0, encWhh0b,
            h0bf[p], h0f, h0bf[1 - p]);
        gru_cell_big<<<ggrid, gblk, 0, stream>>>(
            h0bf[1 - p], encWih1b, encWhh1b, enc_bih1, enc_bhh1,
            h1bf[p], h1f, h1bf[1 - p]);
        p ^= 1;
    }

    // ---- decoder: 30 autoregressive steps ----
    for (int t = 0; t < FUT; ++t) {
        const float* xb;
        int xs;
        if (t == 0) { xb = x_input + (size_t)(THIST - 1) * DDIM; xs = THIST * DDIM; }
        else        { xb = out + (size_t)(t - 1) * DDIM;         xs = FUT * DDIM; }
        gru_cell_small<<<ggrid, gblk, 0, stream>>>(
            xb, xs,
            dec_Wih0, dec_bih0, dec_bhh0, decWhh0b,
            h0bf[p], h0f, h0bf[1 - p]);
        gru_cell_big<<<ggrid, gblk, 0, stream>>>(
            h0bf[1 - p], decWih1b, decWhh1b, dec_bih1, dec_bhh1,
            h1bf[p], h1f, h1bf[1 - p]);
        fc_head<<<BATCH / 8, 256, 0, stream>>>(h1f, fcW, fcb, out, t);
        p ^= 1;
    }
}
